// Backbone_82678120448693
// MI455X (gfx1250) — matrix operand order repaired and checked by reading, not compiled, not run on hardware
//
#include <hip/hip_runtime.h>
#include <hip/hip_bf16.h>

// Fixed problem dims from the reference
#define BDIM   4
#define SDIM   2048
#define DDIM   1024
#define DFFDIM 4096
#define NLAYER 4
#define DHDIM  512

typedef __bf16 bf16_t;
typedef bf16_t v8bf  __attribute__((ext_vector_type(8)));
typedef bf16_t v16bf __attribute__((ext_vector_type(16)));
typedef float  v8f   __attribute__((ext_vector_type(8)));

#if defined(__gfx1250__)
#define USE_ASYNC_LDS 1
#else
#define USE_ASYNC_LDS 0
#endif

__device__ __forceinline__ bf16_t f2bf(float f) {
  union { float f; unsigned u; } x; x.f = f;
  unsigned r = x.u + 0x7fffu + ((x.u >> 16) & 1u);  // round-to-nearest-even
  union { unsigned short s; bf16_t b; } y; y.s = (unsigned short)(r >> 16);
  return y.b;
}

__device__ __forceinline__ float gelu_f(float x) {
  return 0.5f * x * (1.0f + erff(x * 0.70710678118654752f));
}

#if USE_ASYNC_LDS
// gfx1250 async DMA: copy 16B from global straight into LDS, no VGPR transit.
// VDST = VGPR with LDS byte address (low 32 bits of the generic LDS pointer),
// VADDR = 64-bit global address, saddr = off. Tracked by ASYNCcnt.
__device__ __forceinline__ void async_b128(const void* gptr, void* lptr) {
  unsigned loff = (unsigned)(uintptr_t)lptr;
  asm volatile("global_load_async_to_lds_b128 %0, %1, off"
               :: "v"(loff), "v"(gptr)
               : "memory");
}
__device__ __forceinline__ void wait_async0() {
  asm volatile("s_wait_asynccnt 0" ::: "memory");
}
#endif

// One K-step of WMMAs: read this wave's fragments from the staged LDS buffer
// (each lane's 32B operand is contiguous at frag_base + lane*32 -> two
// conflict-free ds_load_b128 per fragment) and run the 2x4 accumulator tiles.
__device__ __forceinline__ void mma_step(const char* sbuf, int wm, int wn,
                                         int lane, v8f acc[2][4]) {
  const char* sa = sbuf;
  const char* sb = sbuf + 8192;
  v16bf a0 = *(const v16bf*)(sa + (wm * 2 + 0) * 1024 + lane * 32);
  v16bf a1 = *(const v16bf*)(sa + (wm * 2 + 1) * 1024 + lane * 32);
  v16bf b0 = *(const v16bf*)(sb + (wn * 4 + 0) * 1024 + lane * 32);
  v16bf b1 = *(const v16bf*)(sb + (wn * 4 + 1) * 1024 + lane * 32);
  v16bf b2 = *(const v16bf*)(sb + (wn * 4 + 2) * 1024 + lane * 32);
  v16bf b3 = *(const v16bf*)(sb + (wn * 4 + 3) * 1024 + lane * 32);

  acc[0][0] = __builtin_amdgcn_wmma_f32_16x16x32_bf16(false, a0, false, b0, (short)0, acc[0][0], false, false);
  acc[0][1] = __builtin_amdgcn_wmma_f32_16x16x32_bf16(false, a0, false, b1, (short)0, acc[0][1], false, false);
  acc[0][2] = __builtin_amdgcn_wmma_f32_16x16x32_bf16(false, a0, false, b2, (short)0, acc[0][2], false, false);
  acc[0][3] = __builtin_amdgcn_wmma_f32_16x16x32_bf16(false, a0, false, b3, (short)0, acc[0][3], false, false);
  acc[1][0] = __builtin_amdgcn_wmma_f32_16x16x32_bf16(false, a1, false, b0, (short)0, acc[1][0], false, false);
  acc[1][1] = __builtin_amdgcn_wmma_f32_16x16x32_bf16(false, a1, false, b1, (short)0, acc[1][1], false, false);
  acc[1][2] = __builtin_amdgcn_wmma_f32_16x16x32_bf16(false, a1, false, b2, (short)0, acc[1][2], false, false);
  acc[1][3] = __builtin_amdgcn_wmma_f32_16x16x32_bf16(false, a1, false, b3, (short)0, acc[1][3], false, false);
}

// ---- GEMM: C[M,N] = A[M,K] (bf16) * W[N,K]^T (bf16) + bias, fused epilogue ---
// EPI 0: outF = v ; EPI 1: outF = v + resid ; EPI 2: outB = bf16(gelu(v))
//
// Block = 256 threads = 8 wave32s; block tile 128x128; wave tile 32x64
// (2x4 WMMA tiles). K stepped by 32, double-buffered through LDS (2 x 16KB).
// Tiles are stored in LDS pre-swizzled into WMMA fragment order; the staging
// chunks are contiguous 16B row pieces (coalesced global access) scattered to
// the fragment-order LDS address. On gfx1250 the staging is done by
// global_load_async_to_lds_b128 (ASYNCcnt-tracked DMA, no VGPR transit).
//
// A operand (16Mx32K): lane l=(hi,lo): bytes[0..15]=A[lo][hi*8+0..7],
//                      bytes[16..31]=A[lo][16+hi*8+0..7]
// B operand (32Kx16N) from W[N,K] row-major: lane l=(hi,lo):
//                      bytes[0..15]=W[lo][hi*8+0..7], bytes[16..31]=W[lo][16+hi*8+0..7]
template <int EPI>
__global__ __launch_bounds__(256) __attribute__((amdgpu_waves_per_eu(2)))
void gemm_bf16_kernel(
    const bf16_t* __restrict__ A, const bf16_t* __restrict__ Bw,
    const float* __restrict__ bias, const float* __restrict__ resid,
    float* __restrict__ outF, bf16_t* __restrict__ outB,
    int M, int N, int K) {
  __shared__ __align__(32) char smem[2][16384];

  const int tid  = threadIdx.x;
  const int lane = tid & 31;
  const int wave = tid >> 5;
  const int wm = wave & 3;   // 4 waves along M (32 rows each)
  const int wn = wave >> 2;  // 2 waves along N (64 cols each)
  const int mblock = blockIdx.y * 128;
  const int nblock = blockIdx.x * 128;

  // --- staging maps: 512 16B chunks per operand tile, 2 per thread ---
  int ldsA[2], ldsB[2];
  const bf16_t* gA[2];
  const bf16_t* gB[2];
#pragma unroll
  for (int c = 0; c < 2; ++c) {
    const int chunk = c * 256 + tid;  // 0..511
    const int row = chunk >> 2;       // 0..127 (A: m row, B: n row)
    const int kc  = chunk & 3;        // which 16B piece of the 32-wide K strip
    const int lo = row & 15, mt = row >> 4;
    const int r = kc * 8;             // k offset: 0, 8, 16, 24
    // A: hi = (r>>3)&1, half = r>>4
    ldsA[c] = mt * 1024 + ((((r >> 3) & 1) * 16 + lo) * 32) + ((r >> 4) * 16);
    // B: hi = (r>>3)&1, half = r>>4
    ldsB[c] = 8192 + mt * 1024 + ((((r >> 3) & 1) * 16 + lo) * 32) + ((r >> 4) * 16);
    gA[c] = A  + (size_t)(mblock + row) * K + r;
    gB[c] = Bw + (size_t)(nblock + row) * K + r;
  }

  v8f acc[2][4];
#pragma unroll
  for (int i = 0; i < 2; ++i)
#pragma unroll
    for (int j = 0; j < 4; ++j) acc[i][j] = {};

#if USE_ASYNC_LDS
  // --- prologue: DMA K-step 0 into buffer 0 ---
#pragma unroll
  for (int c = 0; c < 2; ++c) {
    async_b128(gA[c], smem[0] + ldsA[c]);
    async_b128(gB[c], smem[0] + ldsB[c]);
  }
  int cur = 0;
  for (int kb = 0; kb < K; kb += 32) {
    const bool nxt = (kb + 32) < K;
    wait_async0();    // this wave's DMA into smem[cur] complete
    __syncthreads();  // everyone's DMA complete; prior reads of smem[cur^1] done
    if (nxt) {        // DMA next K-step into the other buffer during the MMAs
#pragma unroll
      for (int c = 0; c < 2; ++c) {
        async_b128(gA[c] + kb + 32, smem[cur ^ 1] + ldsA[c]);
        async_b128(gB[c] + kb + 32, smem[cur ^ 1] + ldsB[c]);
      }
      if (kb + 64 < K) {  // speculative global_prefetch of K-step+2
        __builtin_prefetch((const void*)(gA[0] + kb + 64), 0, 1);
        __builtin_prefetch((const void*)(gB[0] + kb + 64), 0, 1);
      }
    }
    mma_step(smem[cur], wm, wn, lane, acc);
    cur ^= 1;
  }
#else
  // --- fallback: stage via VGPRs ---
  uint4 ra[2], rb[2];
#pragma unroll
  for (int c = 0; c < 2; ++c) {
    ra[c] = *(const uint4*)(gA[c]);
    rb[c] = *(const uint4*)(gB[c]);
  }
#pragma unroll
  for (int c = 0; c < 2; ++c) {
    *(uint4*)(smem[0] + ldsA[c]) = ra[c];
    *(uint4*)(smem[0] + ldsB[c]) = rb[c];
  }
  int cur = 0;
  for (int kb = 0; kb < K; kb += 32) {
    const bool nxt = (kb + 32) < K;
    if (nxt) {
#pragma unroll
      for (int c = 0; c < 2; ++c) {
        ra[c] = *(const uint4*)(gA[c] + kb + 32);
        rb[c] = *(const uint4*)(gB[c] + kb + 32);
      }
    }
    __syncthreads();
    mma_step(smem[cur], wm, wn, lane, acc);
    if (nxt) {
#pragma unroll
      for (int c = 0; c < 2; ++c) {
        *(uint4*)(smem[cur ^ 1] + ldsA[c]) = ra[c];
        *(uint4*)(smem[cur ^ 1] + ldsB[c]) = rb[c];
      }
    }
    cur ^= 1;
  }
#endif

  // C/D layout: VGPR r, lanes 0-15 -> M=r, N=lane; lanes 16-31 -> M=r+8, N=lane-16
  const int lo = lane & 15, hi = lane >> 4;
#pragma unroll
  for (int i = 0; i < 2; ++i) {
#pragma unroll
    for (int j = 0; j < 4; ++j) {
      const int n = nblock + wn * 64 + j * 16 + lo;
      const float bval = bias[n];
#pragma unroll
      for (int r = 0; r < 8; ++r) {
        const int m = mblock + wm * 32 + i * 16 + hi * 8 + r;
        const size_t idx = (size_t)m * N + n;
        float v = acc[i][j][r] + bval;
        if constexpr (EPI == 0) {
          outF[idx] = v;
        } else if constexpr (EPI == 1) {
          outF[idx] = v + resid[idx];
        } else {
          outB[idx] = f2bf(gelu_f(v));
        }
      }
    }
  }
}

// ---- f32 -> (f32 copy?, bf16) conversion ------------------------------------
__global__ __launch_bounds__(256) void cvt_kernel(const float* __restrict__ in,
                                                  float* __restrict__ outF,
                                                  bf16_t* __restrict__ outB,
                                                  size_t n) {
  size_t i = (size_t)blockIdx.x * 256 + threadIdx.x;
  if (i < n) {
    float v = in[i];
    if (outF) outF[i] = v;
    if (outB) outB[i] = f2bf(v);
  }
}

// ---- fused column softmax over S + scores = sum_s gelu(qs*k) ----------------
// grid = B*D blocks, 256 threads; column (b,:,d) of q/k has stride D.
__global__ __launch_bounds__(256) void scores_kernel(const float* __restrict__ q,
                                                     const float* __restrict__ k,
                                                     float* __restrict__ scores) {
  __shared__ float red[256];
  const int col = blockIdx.x;          // b*D + d
  const int b = col >> 10;             // / DDIM
  const int d = col & (DDIM - 1);
  const float* qc = q + (size_t)b * SDIM * DDIM + d;
  const float* kc = k + (size_t)b * SDIM * DDIM + d;
  const int t = threadIdx.x;
  const int PER = SDIM / 256;          // 8

  float qv[SDIM / 256];
  float mx = -3.0e38f;
#pragma unroll
  for (int i = 0; i < PER; ++i) {
    qv[i] = qc[(size_t)(t + i * 256) * DDIM];
    mx = fmaxf(mx, qv[i]);
  }
  red[t] = mx; __syncthreads();
  for (int off = 128; off > 0; off >>= 1) {
    if (t < off) red[t] = fmaxf(red[t], red[t + off]);
    __syncthreads();
  }
  mx = red[0]; __syncthreads();

  float sum = 0.0f;
#pragma unroll
  for (int i = 0; i < PER; ++i) { qv[i] = expf(qv[i] - mx); sum += qv[i]; }
  red[t] = sum; __syncthreads();
  for (int off = 128; off > 0; off >>= 1) {
    if (t < off) red[t] += red[t + off];
    __syncthreads();
  }
  sum = red[0]; __syncthreads();
  const float inv = 1.0f / sum;

  float acc = 0.0f;
#pragma unroll
  for (int i = 0; i < PER; ++i) {
    float kvv = kc[(size_t)(t + i * 256) * DDIM];
    acc += gelu_f(qv[i] * inv * kvv);
  }
  red[t] = acc; __syncthreads();
  for (int off = 128; off > 0; off >>= 1) {
    if (t < off) red[t] += red[t + off];
    __syncthreads();
  }
  if (t == 0) scores[col] = red[0];
}

// ---- att = bf16(v * scores[b,d]) -------------------------------------------
__global__ __launch_bounds__(256) void att_kernel(const float* __restrict__ v,
                                                  const float* __restrict__ scores,
                                                  bf16_t* __restrict__ out) {
  size_t i = (size_t)blockIdx.x * 256 + threadIdx.x;
  const int d = (int)(i & (DDIM - 1));
  const int m = (int)(i >> 10);
  const int b = m >> 11;  // / SDIM
  out[i] = f2bf(v[i] * scores[b * DDIM + d]);
}

// ---- row LayerNorm over D, optional post-add, dual f32/bf16 output ----------
__global__ __launch_bounds__(256) void ln_kernel(const float* __restrict__ in,
                                                 const float* __restrict__ w,
                                                 const float* __restrict__ bsh,
                                                 const float* __restrict__ add_after,
                                                 float* __restrict__ outF,
                                                 bf16_t* __restrict__ outB) {
  __shared__ float red[256];
  const int row = blockIdx.x;
  const float* x = in + (size_t)row * DDIM;
  const int t = threadIdx.x;

  float s = 0.0f, s2 = 0.0f;
  for (int d = t; d < DDIM; d += 256) {
    float v = x[d];
    s += v; s2 += v * v;
  }
  red[t] = s; __syncthreads();
  for (int off = 128; off > 0; off >>= 1) { if (t < off) red[t] += red[t + off]; __syncthreads(); }
  s = red[0]; __syncthreads();
  red[t] = s2; __syncthreads();
  for (int off = 128; off > 0; off >>= 1) { if (t < off) red[t] += red[t + off]; __syncthreads(); }
  s2 = red[0]; __syncthreads();

  const float mean = s * (1.0f / DDIM);
  const float var = s2 * (1.0f / DDIM) - mean * mean;
  const float rstd = rsqrtf(var + 1e-5f);

  for (int d = t; d < DDIM; d += 256) {
    float v = (x[d] - mean) * rstd * w[d] + bsh[d];
    if (add_after) v += add_after[(size_t)row * DDIM + d];
    const size_t idx = (size_t)row * DDIM + d;
    if (outF) outF[idx] = v;
    if (outB) outB[idx] = f2bf(v);
  }
}

extern "C" void kernel_launch(void* const* d_in, const int* in_sizes, int n_in,
                              void* d_out, int out_size, void* d_ws, size_t ws_size,
                              hipStream_t stream) {
  (void)in_sizes; (void)n_in; (void)out_size; (void)ws_size;
  const int M = BDIM * SDIM;              // 8192
  const size_t MD = (size_t)M * DDIM;     // 8,388,608

  const float* season = (const float*)d_in[0];
  const float* trend  = (const float*)d_in[1];
  const float* Wq  = (const float*)d_in[2];
  const float* Wk  = (const float*)d_in[3];
  const float* Wv  = (const float*)d_in[4];
  const float* Wo  = (const float*)d_in[5];
  const float* bq  = (const float*)d_in[6];
  const float* bk  = (const float*)d_in[7];
  const float* bv  = (const float*)d_in[8];
  const float* bo  = (const float*)d_in[9];
  const float* Wc1 = (const float*)d_in[10];
  const float* bc1 = (const float*)d_in[11];
  const float* Wc2 = (const float*)d_in[12];
  const float* bc2 = (const float*)d_in[13];
  const float* ln1w = (const float*)d_in[14];
  const float* ln1b = (const float*)d_in[15];
  const float* ln2w = (const float*)d_in[16];
  const float* ln2b = (const float*)d_in[17];
  const float* normw = (const float*)d_in[18];
  const float* normb = (const float*)d_in[19];
  const float* mW1 = (const float*)d_in[20];
  const float* mb1 = (const float*)d_in[21];
  const float* mW2 = (const float*)d_in[22];
  const float* mb2 = (const float*)d_in[23];
  const float* mW3 = (const float*)d_in[24];
  const float* mb3 = (const float*)d_in[25];
  const float* mlnw = (const float*)d_in[26];
  const float* mlnb = (const float*)d_in[27];

  // ---- workspace carve-up ----
  size_t off = 0;
  auto take = [&](size_t bytes) -> char* {
    char* p = (char*)d_ws + off;
    off += (bytes + 255) & ~(size_t)255;
    return p;
  };
  float*  xF   = (float*)take(MD * 4);          // current x (f32)
  float*  qF   = (float*)take(MD * 4);          // q / pre-LN sum / trend h3
  float*  kF   = (float*)take(MD * 4);          // k / y1B lower half / trend pre
  float*  vF   = (float*)take(MD * 4);          // v / y1B upper half / trend h1b
  bf16_t* xB   = (bf16_t*)take(MD * 2);         // bf16 x / att / trend tb
  float*  scoresF = (float*)take((size_t)BDIM * DDIM * 4);
  bf16_t* WqB  = (bf16_t*)take((size_t)NLAYER * DDIM * DDIM * 2);
  bf16_t* WkB  = (bf16_t*)take((size_t)NLAYER * DDIM * DDIM * 2);
  bf16_t* WvB  = (bf16_t*)take((size_t)NLAYER * DDIM * DDIM * 2);
  bf16_t* WoB  = (bf16_t*)take((size_t)NLAYER * DDIM * DDIM * 2);
  bf16_t* Wc1B = (bf16_t*)take((size_t)NLAYER * DFFDIM * DDIM * 2);
  bf16_t* Wc2B = (bf16_t*)take((size_t)NLAYER * DDIM * DFFDIM * 2);
  bf16_t* mW1B = (bf16_t*)take((size_t)DHDIM * DDIM * 2);
  bf16_t* mW2B = (bf16_t*)take((size_t)DDIM * DHDIM * 2);
  bf16_t* mW3B = (bf16_t*)take((size_t)DDIM * DDIM * 2);
  bf16_t* y1B  = (bf16_t*)kF;   // [M, DFF] bf16 == kF+vF region exactly

  const dim3 blk(256);
  auto cvt = [&](const float* in, float* oF, bf16_t* oB, size_t n) {
    cvt_kernel<<<dim3((unsigned)((n + 255) / 256)), blk, 0, stream>>>(in, oF, oB, n);
  };

  // ---- weight conversions (f32 -> bf16) ----
  cvt(Wq,  nullptr, WqB,  (size_t)NLAYER * DDIM * DDIM);
  cvt(Wk,  nullptr, WkB,  (size_t)NLAYER * DDIM * DDIM);
  cvt(Wv,  nullptr, WvB,  (size_t)NLAYER * DDIM * DDIM);
  cvt(Wo,  nullptr, WoB,  (size_t)NLAYER * DDIM * DDIM);
  cvt(Wc1, nullptr, Wc1B, (size_t)NLAYER * DFFDIM * DDIM);
  cvt(Wc2, nullptr, Wc2B, (size_t)NLAYER * DDIM * DFFDIM);
  cvt(mW1, nullptr, mW1B, (size_t)DHDIM * DDIM);
  cvt(mW2, nullptr, mW2B, (size_t)DDIM * DHDIM);
  cvt(mW3, nullptr, mW3B, (size_t)DDIM * DDIM);

  // ---- season encoder ----
  cvt(season, nullptr, xB, MD);   // xB = bf16(season); f32 residual read from season directly at l=0

  const dim3 gD(DDIM / 128, M / 128);     // N=1024
  const dim3 gFF(DFFDIM / 128, M / 128);  // N=4096
  const dim3 gDH(DHDIM / 128, M / 128);   // N=512

  for (int l = 0; l < NLAYER; ++l) {
    const size_t wo = (size_t)l * DDIM * DDIM;
    const float* xres = (l == 0) ? season : xF;
    // q, k, v projections
    gemm_bf16_kernel<0><<<gD, blk, 0, stream>>>(xB, WqB + wo, bq + l * DDIM, nullptr, qF, nullptr, M, DDIM, DDIM);
    gemm_bf16_kernel<0><<<gD, blk, 0, stream>>>(xB, WkB + wo, bk + l * DDIM, nullptr, kF, nullptr, M, DDIM, DDIM);
    gemm_bf16_kernel<0><<<gD, blk, 0, stream>>>(xB, WvB + wo, bv + l * DDIM, nullptr, vF, nullptr, M, DDIM, DDIM);
    // scores[b,d] = sum_s gelu(softmax_S(q)*k)
    scores_kernel<<<dim3(BDIM * DDIM), blk, 0, stream>>>(qF, kF, scoresF);
    // att = bf16(v * scores)  (overwrites xB, which is no longer needed)
    att_kernel<<<dim3((unsigned)(MD / 256)), blk, 0, stream>>>(vF, scoresF, xB);
    // new_x = att@Wo^T + bo ; pre = x + new_x  -> qF
    gemm_bf16_kernel<1><<<gD, blk, 0, stream>>>(xB, WoB + wo, bo + l * DDIM, xres, qF, nullptr, M, DDIM, DDIM);
    // x = LN1(pre) -> xF (f32) + xB (bf16)
    ln_kernel<<<dim3(M), blk, 0, stream>>>(qF, ln1w + l * DDIM, ln1b + l * DDIM, nullptr, xF, xB);
    // y1 = gelu(x@Wc1^T + bc1) -> bf16 (reuses kF/vF region)
    gemm_bf16_kernel<2><<<gFF, blk, 0, stream>>>(xB, Wc1B + (size_t)l * DFFDIM * DDIM, bc1 + l * DFFDIM,
                                                 nullptr, nullptr, y1B, M, DFFDIM, DDIM);
    // pre = x + y1@Wc2^T + bc2 -> qF
    gemm_bf16_kernel<1><<<gD, blk, 0, stream>>>(y1B, Wc2B + (size_t)l * DDIM * DFFDIM, bc2 + l * DDIM,
                                                xF, qF, nullptr, M, DDIM, DFFDIM);
    // x = LN2(pre)
    ln_kernel<<<dim3(M), blk, 0, stream>>>(qF, ln2w + l * DDIM, ln2b + l * DDIM, nullptr, xF, xB);
  }
  // season_out = LN(x) -> d_out[0 : M*D)
  ln_kernel<<<dim3(M), blk, 0, stream>>>(xF, normw, normb, nullptr, (float*)d_out, nullptr);

  // ---- trend MLP branch (reuses season scratch) ----
  bf16_t* tb  = xB;            // bf16(trend)
  bf16_t* h1b = (bf16_t*)vF;   // [M, DH] bf16
  float*  h3  = qF;            // [M, D]
  float*  preT = kF;           // [M, D]
  cvt(trend, nullptr, tb, MD);
  gemm_bf16_kernel<2><<<gDH, blk, 0, stream>>>(tb, mW1B, mb1, nullptr, nullptr, h1b, M, DHDIM, DDIM);
  gemm_bf16_kernel<0><<<gD,  blk, 0, stream>>>(tb, mW3B, mb3, nullptr, h3, nullptr, M, DDIM, DDIM);
  gemm_bf16_kernel<1><<<gD,  blk, 0, stream>>>(h1b, mW2B, mb2, h3, preT, nullptr, M, DDIM, DHDIM);
  // trend_out = trend + LN(pre) -> d_out[M*D : 2*M*D)
  ln_kernel<<<dim3(M), blk, 0, stream>>>(preT, mlnw, mlnb, trend, (float*)d_out + MD, nullptr);
}
